// DEFORM_ATTEN_19765439496745
// MI455X (gfx1250) — matrix operand order repaired and checked by reading, not compiled, not run on hardware
//
#include <hip/hip_runtime.h>
#include <hip/hip_bf16.h>

// ---------------------------------------------------------------------------
// Deformable attention on MI455X (gfx1250, wave32, WMMA).
// GEMMs run as bf16 V_WMMA_F32_16X16X32_BF16 with f32 accumulate (chip's
// matrix sweet spot; keeps matrix math below the 23.3 TB/s HBM floor).
// Softmax / bilinear sampling stay in fp32.
// Wave tile is 16 x (NT*16): A fragment loaded/converted once per K-step and
// fed to NT back-to-back WMMAs to raise the wmma : (valu+vmem) density.
// ---------------------------------------------------------------------------

typedef __attribute__((ext_vector_type(16))) __bf16 v16bf;
typedef __attribute__((ext_vector_type(8)))  __bf16 v8bf;
typedef __attribute__((ext_vector_type(8)))  float  v8f;

#define NB   8
#define LQS  4096
#define DM   256
#define MH   8
#define PP   4
#define HH   64
#define WW   64
#define DH   32          // DM / MH
#define NQ   (NB * LQS)  // 32768 rows for every GEMM here

// ---------------------------------------------------------------- cvt f32->bf16
__global__ __launch_bounds__(256)
void cvt_f32_bf16(const float* __restrict__ src, __bf16* __restrict__ dst, int n) {
  int i = blockIdx.x * blockDim.x + threadIdx.x;
  if (i < n) dst[i] = (__bf16)src[i];
}

// ---------------------------------------------------------------- WMMA GEMM
// C[M,Nc] = A[M,K] @ W[Nc,K]^T (+bias).  A is f32 (converted on load) or bf16.
// Block: 256 thr = 8 waves; block tile 128 x (NT*16); wave tile 16 x (NT*16).
//
// Fragment layouts per CDNA5 ISA 7.12.2 (wave32):
//  A 16x32 bf16 : lane m=lane&15, half=lane>>4; elems 0..7 = A[m][k0+half*8+i],
//                 elems 8..15 = A[m][k0+16+half*8+i]   (two 16B contiguous runs)
//  B 32x16 bf16 : lane n=lane&15, half=lane>>4; elems 0..7 = W[n][k0+half*8+i],
//                 elems 8..15 = W[n][k0+16+half*8+i]   (two 16B contiguous runs)
//  C/D f32      : reg r -> row r + 8*(lane>>4), col lane&15
template <bool A_IS_F32, int NT>
__global__ __launch_bounds__(256)
void gemm_wmma_bf16(const void* __restrict__ Av, const __bf16* __restrict__ W,
                    const float* __restrict__ bias, float* __restrict__ C,
                    int M, int Nc, int K) {
  const int lane = threadIdx.x & 31;
  const int wave = threadIdx.x >> 5;
  const int row0 = blockIdx.x * 128 + wave * 16;
  const int n0   = blockIdx.y * (NT * 16);
  const int m    = lane & 15;
  const int half = lane >> 4;
  const int nn   = lane & 15;

  v8f acc[NT];
#pragma unroll
  for (int t = 0; t < NT; ++t) acc[t] = (v8f){};

  const float*  Af = (const float*)Av;
  const __bf16* Ab = (const __bf16*)Av;

  const size_t arow = (size_t)(row0 + m) * K;
  const __bf16* wrow[NT];
#pragma unroll
  for (int t = 0; t < NT; ++t) wrow[t] = W + (size_t)(n0 + t * 16 + nn) * K;

  for (int k0 = 0; k0 < K; k0 += 32) {
    // ---- A fragment (one per K-step, shared by all NT wmmas)
    v8bf alo, ahi;
    if (A_IS_F32) {
      const v8f* p0 = (const v8f*)(Af + arow + k0 + half * 8);        // 32B aligned
      const v8f* p1 = (const v8f*)(Af + arow + k0 + 16 + half * 8);
      alo = __builtin_convertvector(p0[0], v8bf);  // v_cvt_pk_bf16_f32
      ahi = __builtin_convertvector(p1[0], v8bf);
    } else {
      const v8bf* p = (const v8bf*)(Ab + arow + k0 + half * 8);       // 16B aligned
      alo = p[0];
      ahi = p[2];  // +16 elements = k0 + 16 + half*8
    }
    v16bf a = __builtin_shufflevector(alo, ahi, 0,1,2,3,4,5,6,7,8,9,10,11,12,13,14,15);

    // ---- B fragments (weight rows contiguous in K: free transpose)
    const int kb = k0 + (half << 3);
#pragma unroll
    for (int t = 0; t < NT; ++t) {
      const v8bf* q = (const v8bf*)(wrow[t] + kb);                    // 16B aligned
      v16bf b = __builtin_shufflevector(q[0], q[2],
                                        0,1,2,3,4,5,6,7,8,9,10,11,12,13,14,15);
      acc[t] = __builtin_amdgcn_wmma_f32_16x16x32_bf16(
          false, a, false, b, (short)0, acc[t], false, false);
    }
  }

  // ---- epilogue: D reg r -> (row0 + r + 8*half, col)
  const size_t base = (size_t)(row0 + 8 * half) * Nc;
#pragma unroll
  for (int t = 0; t < NT; ++t) {
    const int col = n0 + t * 16 + nn;
    const float bv = bias ? bias[col] : 0.0f;
#pragma unroll
    for (int r = 0; r < 8; ++r)
      C[base + (size_t)r * Nc + col] = acc[t][r] + bv;
  }
}

// ---------------------------------------------------------------- sampler
// One wave32 per (n, q, head); lane = channel (DH == 32 == wave size).
// proj row layout: [0..63] = offsets (m*4+p)*2 + xy, [64..95] = attn logits.
__global__ __launch_bounds__(256)
void deform_sample(const float* __restrict__ value,   // [NB*HH*WW, DM]
                   const float* __restrict__ proj,    // [NQ, 96]
                   const float* __restrict__ refpts,  // [NQ, 2]
                   const float* __restrict__ boff,    // [64]
                   const float* __restrict__ ba,      // [32]
                   __bf16* __restrict__ outpre) {     // [NQ, DM]
  const int wave = blockIdx.x * 8 + (threadIdx.x >> 5);
  const int lane = threadIdx.x & 31;
  const int mh = wave & (MH - 1);
  const int q  = (wave >> 3) & (LQS - 1);
  const int n  = wave >> 15;                 // / (MH*LQS)

  const size_t row = (size_t)n * LQS + q;
  const float* pr = proj + row * 96;
  const float rx = refpts[row * 2 + 0];
  const float ry = refpts[row * 2 + 1];

  // softmax over P=4 (fp32, numerically safe)
  float lg[PP];
#pragma unroll
  for (int p = 0; p < PP; ++p) lg[p] = pr[64 + mh * PP + p] + ba[mh * PP + p];
  float mx = fmaxf(fmaxf(lg[0], lg[1]), fmaxf(lg[2], lg[3]));
  float ssum = 0.0f;
#pragma unroll
  for (int p = 0; p < PP; ++p) { lg[p] = __expf(lg[p] - mx); ssum += lg[p]; }
  const float inv = 1.0f / ssum;

  const float* vbase = value + (size_t)n * (HH * WW) * DM + mh * DH + lane;

  float acc = 0.0f;
#pragma unroll
  for (int p = 0; p < PP; ++p) {
    const int oi = (mh * PP + p) * 2;
    const float ox = pr[oi + 0] + boff[oi + 0];
    const float oy = pr[oi + 1] + boff[oi + 1];
    // grid = 2*(ref + off/[W,H]) - 1 ; x = (grid+1)*W/2 - 0.5 = loc*W - 0.5
    const float x = (rx + ox * (1.0f / WW)) * WW - 0.5f;
    const float y = (ry + oy * (1.0f / HH)) * HH - 0.5f;
    const float xf = floorf(x), yf = floorf(y);
    const int x0 = (int)xf, y0 = (int)yf;
    const float wx1 = x - xf, wy1 = y - yf;
    const float wx0 = 1.0f - wx1, wy0 = 1.0f - wy1;

    float sv = 0.0f;
#pragma unroll
    for (int dy = 0; dy < 2; ++dy) {
#pragma unroll
      for (int dx = 0; dx < 2; ++dx) {
        const int xi = x0 + dx, yi = y0 + dy;
        if (xi >= 0 && xi < WW && yi >= 0 && yi < HH) {
          const float w = (dx ? wx1 : wx0) * (dy ? wy1 : wy0);
          sv += vbase[(size_t)(yi * WW + xi) * DM] * w;   // coalesced 128B / wave
        }
      }
    }
    acc += sv * lg[p] * inv;
  }
  outpre[row * DM + mh * DH + lane] = (__bf16)acc;
}

// ---------------------------------------------------------------- launch
extern "C" void kernel_launch(void* const* d_in, const int* in_sizes, int n_in,
                              void* d_out, int out_size, void* d_ws, size_t ws_size,
                              hipStream_t stream) {
  (void)in_sizes; (void)n_in; (void)out_size; (void)ws_size;
  const float* query  = (const float*)d_in[0];
  const float* refpts = (const float*)d_in[1];
  const float* inflat = (const float*)d_in[2];
  // d_in[3] = H, d_in[4] = W (compile-time 64 here)
  const float* Wv   = (const float*)d_in[5];
  const float* bv   = (const float*)d_in[6];
  const float* Woff = (const float*)d_in[7];
  const float* boff = (const float*)d_in[8];
  const float* Wa   = (const float*)d_in[9];
  const float* ba   = (const float*)d_in[10];
  const float* Wo   = (const float*)d_in[11];
  const float* bo   = (const float*)d_in[12];
  float* out = (float*)d_out;

  // workspace carve-out (~60.3 MB)
  char* p = (char*)d_ws;
  __bf16* Wv_b    = (__bf16*)p; p += (size_t)DM * DM * sizeof(__bf16);
  __bf16* Wproj_b = (__bf16*)p; p += (size_t)96 * DM * sizeof(__bf16);
  __bf16* Wo_b    = (__bf16*)p; p += (size_t)DM * DM * sizeof(__bf16);
  float*  value   = (float*)p;  p += (size_t)NB * HH * WW * DM * sizeof(float);
  float*  proj    = (float*)p;  p += (size_t)NQ * 96 * sizeof(float);
  __bf16* outpre  = (__bf16*)p; p += (size_t)NQ * DM * sizeof(__bf16);

  // weights -> bf16 (tiny)
  cvt_f32_bf16<<<(DM * DM + 255) / 256, 256, 0, stream>>>(Wv, Wv_b, DM * DM);
  cvt_f32_bf16<<<(64 * DM + 255) / 256, 256, 0, stream>>>(Woff, Wproj_b, 64 * DM);
  cvt_f32_bf16<<<(32 * DM + 255) / 256, 256, 0, stream>>>(Wa, Wproj_b + 64 * DM, 32 * DM);
  cvt_f32_bf16<<<(DM * DM + 255) / 256, 256, 0, stream>>>(Wo, Wo_b, DM * DM);

  dim3 blk(256);
  // value projection: [NQ,256] = input_flatten(f32) @ Wv^T + bv   (NT=4: 16x64 tile)
  gemm_wmma_bf16<true, 4><<<dim3(NQ / 128, DM / 64), blk, 0, stream>>>(
      inflat, Wv_b, bv, value, NQ, DM, DM);
  // fused offsets+attn projection: [NQ,96] = query(f32) @ [Woff;Wa]^T (NT=3: 16x48 tile)
  gemm_wmma_bf16<true, 3><<<dim3(NQ / 128, 96 / 48), blk, 0, stream>>>(
      query, Wproj_b, nullptr, proj, NQ, 96, DM);
  // softmax + bilinear sample + head merge -> bf16 [NQ,256]
  deform_sample<<<(NQ * MH) / 8, blk, 0, stream>>>(value, proj, refpts, boff, ba, outpre);
  // output projection: [NQ,256] = outpre(bf16) @ Wo^T + bo        (NT=4: 16x64 tile)
  gemm_wmma_bf16<false, 4><<<dim3(NQ / 128, DM / 64), blk, 0, stream>>>(
      outpre, Wo_b, bo, out, NQ, DM, DM);
}
